// RelationClassificationMultiInstance_13846974562934
// MI455X (gfx1250) — matrix operand order repaired and checked by reading, not compiled, not run on hardware
//
#include <hip/hip_runtime.h>

typedef __attribute__((ext_vector_type(16))) _Float16 v16h;
typedef __attribute__((ext_vector_type(8)))  _Float16 v8h;
typedef __attribute__((ext_vector_type(8)))  float    v8f;

#define NEGV (-1.0e30f)

// Problem dims (from reference setup_inputs)
constexpr int Bz = 2, Sq = 256, Hd = 768, Mm = 64, Pp = 128, MPp = 512, Kp = 16;
constexpr int META = 32, Rr = 97, RPAD = 128;
constexpr int K1  = 5 * Hd + 2 * META;   // 3904 = 122*32
constexpr int K2  = Hd + 2 * META;       // 832  = 26*32
constexpr int NR1 = Bz * MPp;            // 1024
constexpr int NR2 = Bz * Pp;             // 256

// ---------------------------------------------------------------------------
// f32 -> f16 conversion (optionally zero-padding the tail)
// ---------------------------------------------------------------------------
__global__ void __launch_bounds__(256) cvt_f16(const float* __restrict__ src,
                                               _Float16* __restrict__ dst,
                                               int nvalid, int ntotal) {
  int i = blockIdx.x * 256 + threadIdx.x;
  if (i < ntotal) dst[i] = (i < nvalid) ? (_Float16)src[i] : (_Float16)0.0f;
}

// ---------------------------------------------------------------------------
// Build local = [rel_ctx | mention0 | mention1 | entity_pair | tok_d | sent_d]
// One block (256 threads) per (b, mp) row; masks cached in LDS.
// ---------------------------------------------------------------------------
__global__ void __launch_bounds__(256) build_local(
    const float* __restrict__ ep_reprs,   // (B,P,2H)
    const float* __restrict__ hptr,       // (B,S,H)
    const float* __restrict__ ment,       // (B,M,H)
    const int*   __restrict__ mp_ep,      // (B,MP)
    const int*   __restrict__ m_pairs,    // (B,MP,2)
    const int*   __restrict__ ctx_masks,  // (B,MP,S)
    const int*   __restrict__ tok_idx,    // (B,MP)
    const int*   __restrict__ sent_idx,   // (B,MP)
    const float* __restrict__ tok_emb,    // (TD,32)
    const float* __restrict__ sent_emb,   // (SD,32)
    _Float16* __restrict__ localF)        // (NR1,K1)
{
  const int r  = blockIdx.x;              // 0..NR1-1
  const int b  = r / MPp;
  const int mp = r - b * MPp;
  const int t  = threadIdx.x;             // 0..255 (== S)

  __shared__ int smask[Sq];
  smask[t] = ctx_masks[((size_t)b * MPp + mp) * Sq + t];
  __syncthreads();

  _Float16* lrow = localF + (size_t)r * K1;
  const float* hb = hptr + (size_t)b * Sq * Hd;

  // masked max over sequence positions (3 h-dims per thread)
  const int j0 = t, j1 = t + 256, j2 = t + 512;
  float m0 = -3.0e38f, m1 = -3.0e38f, m2 = -3.0e38f;
  bool anyv = false;
  for (int s = 0; s < Sq; ++s) {
    if (smask[s]) {
      anyv = true;
      const float* hs = hb + (size_t)s * Hd;
      m0 = fmaxf(m0, hs[j0]);
      m1 = fmaxf(m1, hs[j1]);
      m2 = fmaxf(m2, hs[j2]);
    }
  }
  lrow[j0] = (_Float16)(anyv ? m0 : 0.0f);
  lrow[j1] = (_Float16)(anyv ? m1 : 0.0f);
  lrow[j2] = (_Float16)(anyv ? m2 : 0.0f);

  // mention pair reprs
  const int pm0 = m_pairs[((size_t)b * MPp + mp) * 2 + 0];
  const int pm1 = m_pairs[((size_t)b * MPp + mp) * 2 + 1];
  const float* mr0 = ment + ((size_t)b * Mm + pm0) * Hd;
  const float* mr1 = ment + ((size_t)b * Mm + pm1) * Hd;
  for (int j = t; j < Hd; j += 256) {
    lrow[Hd + j]     = (_Float16)mr0[j];
    lrow[2 * Hd + j] = (_Float16)mr1[j];
  }

  // entity-pair repr
  const int pep = mp_ep[(size_t)b * MPp + mp];
  const float* epr = ep_reprs + ((size_t)b * Pp + pep) * (2 * Hd);
  for (int j = t; j < 2 * Hd; j += 256) lrow[3 * Hd + j] = (_Float16)epr[j];

  // distance embeddings
  if (t < META) {
    lrow[5 * Hd + t]        = (_Float16)tok_emb [(size_t)tok_idx [(size_t)b * MPp + mp] * META + t];
    lrow[5 * Hd + META + t] = (_Float16)sent_emb[(size_t)sent_idx[(size_t)b * MPp + mp] * META + t];
  }
}

// ---------------------------------------------------------------------------
// Build lr_te = [max_k pair_repr[gather] | type_emb x2], f16 (NR2 x K2)
// ---------------------------------------------------------------------------
__global__ void __launch_bounds__(256) build_lrte(
    const float* __restrict__ pair_repr,  // (NR1,H) f32
    const int*   __restrict__ ep_mp,      // (B,P,K)
    const int*   __restrict__ pair_masks, // (B,P,K)
    const int*   __restrict__ ent_types,  // (B,P,2)
    const float* __restrict__ type_emb,   // (ET,32)
    _Float16* __restrict__ lrte)          // (NR2,K2)
{
  const int r = blockIdx.x;               // 0..NR2-1
  const int b = r / Pp;
  const int p = r - b * Pp;
  const int t = threadIdx.x;

  __shared__ int sidx[Kp];
  __shared__ int smsk[Kp];
  if (t < Kp) {
    sidx[t] = ep_mp     [((size_t)b * Pp + p) * Kp + t];
    smsk[t] = pair_masks[((size_t)b * Pp + p) * Kp + t];
  }
  __syncthreads();

  _Float16* orow = lrte + (size_t)r * K2;
  for (int j = t; j < Hd; j += 256) {
    float mx = -3.0e38f;
    #pragma unroll
    for (int k = 0; k < Kp; ++k) {
      float v = pair_repr[((size_t)b * MPp + sidx[k]) * Hd + j];
      if (!smsk[k]) v += NEGV;            // matches reference: + NEG when mask==0
      mx = fmaxf(mx, v);
    }
    orow[j] = (_Float16)mx;
  }
  if (t < 2 * META) {
    const int e = ent_types[((size_t)b * Pp + p) * 2 + (t >> 5)];
    orow[Hd + t] = (_Float16)type_emb[(size_t)e * META + (t & 31)];
  }
}

// ---------------------------------------------------------------------------
// WMMA GEMM, block-cooperative with CDNA5 async global->LDS W staging.
//   out[M][LDO] = A[M][K](f16) * W[N][K]^T(f16) + bias, opt ReLU.
// Block = 8 waves = 128 rows x 64 cols. W tile (64x32 f16 = 4KB) staged in LDS
// via global_load_async_to_lds_b128 (ASYNCcnt), double buffered. K-loop is
// manually unrolled 2x with ping-pong A fragments (a0/a1) so no register
// copies are needed; all 4 B fragments are ds-loaded before the 4 WMMAs so
// there is one dscnt wait per k-step and the WMMA chain issues back-to-back.
// LDS tile rows padded to 80B to skew bank access. W must have >= NCOLS rows
// of K f16 (caller zero-pads beyond valid rows).
// Fragment layouts used here (wave32; A/B use contiguous k, not ISA table order):
//   A 16x32 f16 : lane L (half=L>>4, m=L&15): elem j -> k = kb + 16*half + j
//                 (contiguous, matching B)
//   B 32x16 f16 : lane L (n=L&15): elem j -> k = kb + 16*half + j (contiguous)
//   C/D 16x16 f32: elem v -> row = v + 8*half, col = L&15
// ---------------------------------------------------------------------------
template<int MROWS, int NCOLS, int KDIM, int NGUARD, int LDO, bool RELU, bool OUTF16>
__global__ void __launch_bounds__(256) wmma_gemm_async(
    const _Float16* __restrict__ A,
    const _Float16* __restrict__ W,
    const float*    __restrict__ bias,
    float*          __restrict__ o32,
    _Float16*       __restrict__ o16)
{
  constexpr int BM = 128, BN = 64, KSTEP = 32;
  constexpr int NKS = KDIM / KSTEP;
  static_assert(NKS % 2 == 0, "K pipeline assumes an even number of k-steps");
  constexpr int LDT = 40;                       // halves per LDS tile row (80B)
  constexpr unsigned TILE_BYTES = BN * LDT * 2; // 5120B per buffer
  __shared__ _Float16 wtile[2][BN * LDT];

  constexpr int MBLK = MROWS / BM;
  const int mblk = blockIdx.x % MBLK;
  const int nblk = blockIdx.x / MBLK;
  const int row0 = mblk * BM + (int)(threadIdx.x >> 5) * 16;
  const int n0   = nblk * BN;

  const int lane = threadIdx.x & 31;
  const int half = lane >> 4;
  const int sub  = lane & 15;
  const int t    = threadIdx.x;

  // cooperative async copy: thread t moves 16B/stage (tile row t/4, 16B chunk t%4)
  const int crow  = t >> 2;
  const int cpart = t & 3;
  const _Float16* wsrc = W + (size_t)(n0 + crow) * KDIM + cpart * 8;
  // LDS byte address (flat shared addr low 32 bits == LDS offset)
  const unsigned ldsoff0 =
      (unsigned)(unsigned long long)(const void*)&wtile[0][crow * LDT + cpart * 8];

  auto stage = [&](int ks, int buf) {
    unsigned long long ga = (unsigned long long)(wsrc + (size_t)ks * KSTEP);
    unsigned loff = ldsoff0 + (unsigned)buf * TILE_BYTES;
    asm volatile("global_load_async_to_lds_b128 %0, %1, off"
                 :: "v"(loff), "v"(ga) : "memory");
  };

  const _Float16* Arow = A + (size_t)(row0 + sub) * KDIM;
  auto loadA = [&](int ks, v16h& af) {
    v8h alo = *(const v8h*)(Arow + ks * KSTEP + 16 * half);
    v8h ahi = *(const v8h*)(Arow + ks * KSTEP + 16 * half + 8);
    #pragma unroll
    for (int j = 0; j < 8; ++j) { af[j] = alo[j]; af[8 + j] = ahi[j]; }
  };

  // per-buffer LDS read bases (loop-invariant; nt strides become immediates)
  const _Float16* wbase0 = &wtile[0][sub * LDT + half * 16];
  const _Float16* wbase1 = &wtile[1][sub * LDT + half * 16];

  v8f acc[4] = {};

  auto compute = [&](const _Float16* wbase, const v16h& af) {
    v16h bf[4];
    #pragma unroll
    for (int nt = 0; nt < 4; ++nt) {           // 8 ds_load_b128, clause-able
      const _Float16* wb = wbase + nt * (16 * LDT);
      v8h blo = *(const v8h*)(wb);
      v8h bhi = *(const v8h*)(wb + 8);
      #pragma unroll
      for (int j = 0; j < 8; ++j) { bf[nt][j] = blo[j]; bf[nt][8 + j] = bhi[j]; }
    }
    #pragma unroll
    for (int nt = 0; nt < 4; ++nt)             // 4 WMMAs back-to-back
      acc[nt] = __builtin_amdgcn_wmma_f32_16x16x32_f16(
          /*neg_a=*/false, af, /*neg_b=*/false, bf[nt],
          /*c_mod=*/(short)0, acc[nt], /*reuse_a=*/false, /*reuse_b=*/false);
  };

  v16h a0, a1;
  stage(0, 0);                                  // asynccnt: 1
  loadA(0, a0);

  for (int ks = 0; ks < NKS; ks += 2) {
    // ---- even half-step: consumes buf0 / a0 -----------------------------
    stage(ks + 1, 1);                                        // cnt -> 2
    asm volatile("s_wait_asynccnt 0x1" ::: "memory");        // buf0 landed
    __syncthreads();
    loadA(ks + 1, a1);
    compute(wbase0, a0);
    __syncthreads();   // all waves done with buf0 before it is restaged

    // ---- odd half-step: consumes buf1 / a1 ------------------------------
    const bool more = (ks + 2 < NKS);
    if (more) {
      stage(ks + 2, 0);                                      // cnt -> 2
      asm volatile("s_wait_asynccnt 0x1" ::: "memory");      // buf1 landed
    } else {
      asm volatile("s_wait_asynccnt 0x0" ::: "memory");
    }
    __syncthreads();
    if (more) loadA(ks + 2, a0);
    compute(wbase1, a1);
    __syncthreads();   // all waves done with buf1 before it is restaged
  }

  #pragma unroll
  for (int nt = 0; nt < 4; ++nt) {
    const int col = n0 + nt * 16 + sub;
    if (NGUARD < NCOLS && col >= NGUARD) continue;
    const float bv = bias[col];
    #pragma unroll
    for (int v = 0; v < 8; ++v) {
      const int rowm = row0 + v + 8 * half;
      float val = acc[nt][v] + bv;
      if (RELU) val = fmaxf(val, 0.0f);
      if constexpr (OUTF16) o16[(size_t)rowm * LDO + col] = (_Float16)val;
      else                  o32[(size_t)rowm * LDO + col] = val;
    }
  }
}

// ---------------------------------------------------------------------------
extern "C" void kernel_launch(void* const* d_in, const int* in_sizes, int n_in,
                              void* d_out, int out_size, void* d_ws, size_t ws_size,
                              hipStream_t stream) {
  const float* ep_reprs   = (const float*)d_in[0];
  const float* hbuf       = (const float*)d_in[1];
  const float* ment       = (const float*)d_in[2];
  const int*   ep_mp      = (const int*)  d_in[3];
  const int*   mp_ep      = (const int*)  d_in[4];
  const int*   m_pairs    = (const int*)  d_in[5];
  const int*   ctx_masks  = (const int*)  d_in[6];
  const int*   pair_masks = (const int*)  d_in[7];
  const int*   tok_idx    = (const int*)  d_in[8];
  const int*   sent_idx   = (const int*)  d_in[9];
  const int*   ent_types  = (const int*)  d_in[10];
  const float* pair_w     = (const float*)d_in[11];
  const float* pair_b     = (const float*)d_in[12];
  const float* rel_w      = (const float*)d_in[13];
  const float* rel_b      = (const float*)d_in[14];
  const float* clf_w      = (const float*)d_in[15];
  const float* clf_b      = (const float*)d_in[16];
  const float* tok_emb    = (const float*)d_in[17];
  const float* sent_emb   = (const float*)d_in[18];
  const float* type_emb   = (const float*)d_in[19];

  // workspace carve-out (~19.4 MB total)
  char*  ws  = (char*)d_ws;
  size_t off = 0;
  auto take = [&](size_t bytes) -> char* {
    char* p = ws + off;
    off = (off + bytes + 255) & ~(size_t)255;
    return p;
  };
  _Float16* localF  = (_Float16*)take((size_t)NR1  * K1 * 2);  // 8.0 MB
  _Float16* pairwF  = (_Float16*)take((size_t)Hd   * K1 * 2);  // 6.0 MB
  _Float16* relwF   = (_Float16*)take((size_t)Hd   * K2 * 2);  // 1.3 MB
  _Float16* clfwF   = (_Float16*)take((size_t)RPAD * Hd * 2);  // 0.2 MB (97->128 zero-pad)
  float*    pairRep = (float*)   take((size_t)NR1  * Hd * 4);  // 3.1 MB
  _Float16* lrteF   = (_Float16*)take((size_t)NR2  * K2 * 2);  // 0.4 MB
  _Float16* relRep  = (_Float16*)take((size_t)NR2  * Hd * 2);  // 0.4 MB

  // 1) weight conversions to f16 (classifier zero-padded to 128 rows)
  cvt_f16<<<(Hd * K1 + 255) / 256, 256, 0, stream>>>(pair_w, pairwF, Hd * K1, Hd * K1);
  cvt_f16<<<(Hd * K2 + 255) / 256, 256, 0, stream>>>(rel_w,  relwF,  Hd * K2, Hd * K2);
  cvt_f16<<<(RPAD * Hd + 255) / 256, 256, 0, stream>>>(clf_w, clfwF, Rr * Hd, RPAD * Hd);

  // 2) fused masked-max + gather + concat -> local (f16)
  build_local<<<NR1, 256, 0, stream>>>(ep_reprs, hbuf, ment, mp_ep, m_pairs,
                                       ctx_masks, tok_idx, sent_idx,
                                       tok_emb, sent_emb, localF);

  // 3) GEMM1: pair_repr[1024,768] = local[1024,3904] x pair_w^T + pair_b
  //    (1024/128) x (768/64) = 96 blocks
  wmma_gemm_async<NR1, Hd, K1, Hd, Hd, false, false>
      <<<96, 256, 0, stream>>>(localF, pairwF, pair_b, pairRep, nullptr);

  // 4) segment-max gather + type embeddings -> lrte (f16, 256x832)
  build_lrte<<<NR2, 256, 0, stream>>>(pairRep, ep_mp, pair_masks, ent_types,
                                      type_emb, lrteF);

  // 5) GEMM2: rel_repr[256,768] = relu(lrte x rel_w^T + rel_b), f16 out
  //    (256/128) x (768/64) = 24 blocks
  wmma_gemm_async<NR2, Hd, K2, Hd, Hd, true, true>
      <<<24, 256, 0, stream>>>(lrteF, relwF, rel_b, nullptr, relRep);

  // 6) GEMM3: logits[256,97] = rel_repr x clf_w^T + clf_b (N padded 97->128)
  //    (256/128) x (128/64) = 4 blocks
  wmma_gemm_async<NR2, 128, Hd, Rr, Rr, false, false>
      <<<4, 256, 0, stream>>>(relRep, clfwF, clf_b, (float*)d_out, nullptr);
}
